// StudentAgent_10883447128701
// MI455X (gfx1250) — matrix operand order repaired and checked by reading, not compiled, not run on hardware
//
#include <hip/hip_runtime.h>

typedef __attribute__((ext_vector_type(16))) _Float16 v16h;
typedef __attribute__((ext_vector_type(8)))  float    v8f;

#define TSTEPS 256
#define BATCH  1024
#define OBSD   128
#define LSD    128
#define HIDD   128
#define TB     (TSTEPS * BATCH)

// ---------- WMMA fragment helpers (wave32, 16x16x32 f16 shapes) ----------
// A 16x32 f16: lanes 0-15 -> M=lane, K in {0..15}; lanes 16-31 -> M=lane-16, K in {16..31}
__device__ __forceinline__ v16h load_A_g_f32(const float* __restrict__ base, int ldk, int lane) {
  const int m = lane & 15;
  const int kb = (lane & 16) ? 16 : 0;
  const float* row = base + (size_t)m * ldk;
  v16h a;
#pragma unroll
  for (int e = 0; e < 16; ++e) {
    int k = kb + e;
    a[e] = (_Float16)row[k];
  }
  return a;
}

__device__ __forceinline__ v16h load_A_f16(const _Float16* __restrict__ base, int ldk, int lane) {
  const int m = lane & 15;
  const int kb = (lane & 16) ? 16 : 0;
  const _Float16* row = base + (size_t)m * ldk;
  v16h a;
#pragma unroll
  for (int e = 0; e < 16; ++e) {
    int k = kb + e;
    a[e] = row[k];
  }
  return a;
}

// B 32x16 f16 gathered from row-major W[N][K] (i.e. B = W^T tile):
// lanes 0-15 hold K=k0..k0+15, lanes 16-31 hold K=k0+16..k0+31; N = lane&15
__device__ __forceinline__ v16h load_B_W_f16(const _Float16* __restrict__ w, int ldk,
                                             int n0, int k0, int lane) {
  const int n = lane & 15;
  const int kb = (lane & 16) ? 16 : 0;
  const _Float16* row = w + (size_t)(n0 + n) * ldk + (k0 + kb);
  v16h b;
#pragma unroll
  for (int e = 0; e < 16; ++e) b[e] = row[e];
  return b;
}

__device__ __forceinline__ v8f wmma_f16(v16h a, v16h b, v8f c) {
  return __builtin_amdgcn_wmma_f32_16x16x32_f16(false, a, false, b, (short)0, c, false, false);
}

// ---------- prep: all GEMM weights -> f16 once; pack (Wa,Wc,0) head ----------
__global__ __launch_bounds__(256)
void prep_kernel(const float* __restrict__ Whh, _Float16* __restrict__ Whh16,
                 const float* __restrict__ W1,  _Float16* __restrict__ W1h,
                 const float* __restrict__ W2,  _Float16* __restrict__ W2h,
                 const float* __restrict__ W3,  _Float16* __restrict__ W3h,
                 const float* __restrict__ Wih, _Float16* __restrict__ Wih16,
                 const float* __restrict__ Wa, const float* __restrict__ ba,
                 const float* __restrict__ Wc, const float* __restrict__ bc,
                 _Float16* __restrict__ Whead, float* __restrict__ bhead) {
  int i = blockIdx.x * 256 + threadIdx.x;
  if (i < 512 * 128) {
    Whh16[i] = (_Float16)Whh[i];
    Wih16[i] = (_Float16)Wih[i];
  }
  if (i < 128 * 128) W1h[i] = (_Float16)W1[i];
  if (i < 32 * 128)  W2h[i] = (_Float16)W2[i];
  if (i < 128 * 32)  W3h[i] = (_Float16)W3[i];
  if (i < 32 * 128) {
    int n = i >> 7, k = i & 127;
    float v = 0.f;
    if (n < 18)       v = Wa[n * 128 + k];
    else if (n == 18) v = Wc[k];
    Whead[i] = (_Float16)v;
  }
  if (i < 32) {
    float v = 0.f;
    if (i < 18)       v = ba[i];
    else if (i == 18) v = bc[0];
    bhead[i] = v;
  }
}

// ---------- fused 3-layer MLP encoder: x[TB,128] -> hid[TB,128] f16 ----------
__global__ __launch_bounds__(256)
void encoder_kernel(const float* __restrict__ x,
                    const _Float16* __restrict__ W1h, const float* __restrict__ b1,
                    const _Float16* __restrict__ W2h, const float* __restrict__ b2,
                    const _Float16* __restrict__ W3h, const float* __restrict__ b3,
                    _Float16* __restrict__ hid) {
  __shared__ _Float16 sH1[8][16][128];  // per-wave staging, 32KB
  __shared__ _Float16 sH2[8][16][32];   // per-wave staging, 8KB
  const int lane = threadIdx.x & 31;
  const int wv = threadIdx.x >> 5;
  const int row0 = blockIdx.x * 128 + wv * 16;
  const int col = lane & 15;
  const int rb = (lane & 16) ? 8 : 0;

  { // layer 1: 128 -> 128, relu
    v16h a[4];
#pragma unroll
    for (int kt = 0; kt < 4; ++kt)
      a[kt] = load_A_g_f32(x + (size_t)row0 * OBSD + kt * 32, OBSD, lane);
#pragma unroll
    for (int nt = 0; nt < 8; ++nt) {
      v8f acc = {};
#pragma unroll
      for (int kt = 0; kt < 4; ++kt)
        acc = wmma_f16(a[kt], load_B_W_f16(W1h, OBSD, nt * 16, kt * 32, lane), acc);
      float bv = b1[nt * 16 + col];
#pragma unroll
      for (int v = 0; v < 8; ++v) {
        float r = acc[v] + bv;
        sH1[wv][rb + v][nt * 16 + col] = (_Float16)(r > 0.f ? r : 0.f);
      }
    }
  }
  { // layer 2: 128 -> 32, relu
    v16h a[4];
#pragma unroll
    for (int kt = 0; kt < 4; ++kt)
      a[kt] = load_A_f16(&sH1[wv][0][kt * 32], 128, lane);
#pragma unroll
    for (int nt = 0; nt < 2; ++nt) {
      v8f acc = {};
#pragma unroll
      for (int kt = 0; kt < 4; ++kt)
        acc = wmma_f16(a[kt], load_B_W_f16(W2h, 128, nt * 16, kt * 32, lane), acc);
      float bv = b2[nt * 16 + col];
#pragma unroll
      for (int v = 0; v < 8; ++v) {
        float r = acc[v] + bv;
        sH2[wv][rb + v][nt * 16 + col] = (_Float16)(r > 0.f ? r : 0.f);
      }
    }
  }
  { // layer 3: 32 -> 128, relu -> global f16
    v16h a = load_A_f16(&sH2[wv][0][0], 32, lane);
#pragma unroll
    for (int nt = 0; nt < 8; ++nt) {
      v8f acc = {};
      acc = wmma_f16(a, load_B_W_f16(W3h, 32, nt * 16, 0, lane), acc);
      float bv = b3[nt * 16 + col];
#pragma unroll
      for (int v = 0; v < 8; ++v) {
        float r = acc[v] + bv;
        hid[(size_t)(row0 + rb + v) * LSD + nt * 16 + col] = (_Float16)(r > 0.f ? r : 0.f);
      }
    }
  }
}

// ---------- xg[TB,512] = hid @ Wih^T + (bih + bhh), precomputed input gates ----------
__global__ __launch_bounds__(256)
void xgates_kernel(const _Float16* __restrict__ hid, const _Float16* __restrict__ Wih16,
                   const float* __restrict__ bih, const float* __restrict__ bhh,
                   _Float16* __restrict__ xg) {
  const int lane = threadIdx.x & 31;
  const int wv = threadIdx.x >> 5;
  const int row0 = blockIdx.x * 128 + wv * 16;
  const int cbase = blockIdx.y * 128;
  const int col = lane & 15;
  const int rb = (lane & 16) ? 8 : 0;
  v16h a[4];
#pragma unroll
  for (int kt = 0; kt < 4; ++kt)
    a[kt] = load_A_f16(hid + (size_t)row0 * LSD + kt * 32, LSD, lane);
#pragma unroll
  for (int nt = 0; nt < 8; ++nt) {
    const int n0 = cbase + nt * 16;
    v8f acc = {};
#pragma unroll
    for (int kt = 0; kt < 4; ++kt)
      acc = wmma_f16(a[kt], load_B_W_f16(Wih16, LSD, n0, kt * 32, lane), acc);
    float bv = bih[n0 + col] + bhh[n0 + col];
#pragma unroll
    for (int v = 0; v < 8; ++v)
      xg[(size_t)(row0 + rb + v) * 512 + n0 + col] = (_Float16)(acc[v] + bv);
  }
}

// ---------- recurrent LSTM: 64 blocks, each privately owns 16 batch rows for all T ----------
// Per step: async-DMA the 16x512 f16 xg tile into LDS (overlapped with h@Whh^T WMMAs),
// h/c state lives in registers (thread owns col j = tid&127, rows (tid>>7)+2k).
__global__ __launch_bounds__(256)
void lstm_kernel(const float* __restrict__ h0, const float* __restrict__ c0,
                 const float* __restrict__ done, const _Float16* __restrict__ Whh16,
                 const _Float16* __restrict__ xg, _Float16* __restrict__ hs) {
  __shared__ float    sGates[16][512];  // 32KB: h @ Whh^T accumulators
  __shared__ _Float16 sXg[16][512];     // 16KB: async-staged input gates
  __shared__ _Float16 sH16[16][128];    // 4KB:  f16 copy of h for WMMA
  const int tid = threadIdx.x;
  const int lane = tid & 31;
  const int wv = tid >> 5;
  const int r0 = blockIdx.x * 16;
  const int col = lane & 15;
  const int rb = (lane & 16) ? 8 : 0;
  const int j = tid & 127;       // this thread's hidden column
  const int rbase = tid >> 7;    // rows rbase, rbase+2, ..., rbase+14

  float hreg[8], creg[8];
#pragma unroll
  for (int k = 0; k < 8; ++k) {
    int r = rbase + 2 * k;
    hreg[k] = h0[(size_t)(r0 + r) * HIDD + j];
    creg[k] = c0[(size_t)(r0 + r) * HIDD + j];
  }
  const unsigned ldsXg = (unsigned)(size_t)(void*)&sXg[0][0];

  for (int t = 0; t < TSTEPS; ++t) {
    // 1) episode-reset mask, publish f16 h to LDS
#pragma unroll
    for (int k = 0; k < 8; ++k) {
      int r = rbase + 2 * k;
      float m = 1.0f - done[(size_t)t * BATCH + r0 + r];
      hreg[k] *= m;
      creg[k] *= m;
      sH16[r][j] = (_Float16)hreg[k];
    }
    // 2) kick off async DMA of this step's xg tile (16x512 f16, contiguous 16KB)
    {
      const char* gsrc = (const char*)(xg + ((size_t)t * BATCH + r0) * 512);
#pragma unroll
      for (int i = 0; i < 4; ++i) {
        unsigned boff = (unsigned)(tid + i * 256) * 16u;
        asm volatile("global_load_async_to_lds_b128 %0, %1, off"
                     :: "v"(ldsXg + boff), "v"((const void*)(gsrc + boff))
                     : "memory");
      }
    }
    __syncthreads();

    // 3) gates_acc = h @ Whh^T; wave wv owns 64 gate columns
    v16h a[4];
#pragma unroll
    for (int kt = 0; kt < 4; ++kt)
      a[kt] = load_A_f16(&sH16[0][kt * 32], 128, lane);
#pragma unroll
    for (int nt = 0; nt < 4; ++nt) {
      const int n0 = wv * 64 + nt * 16;
      v8f acc = {};
#pragma unroll
      for (int kt = 0; kt < 4; ++kt)
        acc = wmma_f16(a[kt], load_B_W_f16(Whh16, HIDD, n0, kt * 32, lane), acc);
#pragma unroll
      for (int v = 0; v < 8; ++v)
        sGates[rb + v][n0 + col] = acc[v];
    }
    if (t + 1 < TSTEPS)  // warm L2 for the next step's tile
      __builtin_prefetch(xg + ((size_t)(t + 1) * BATCH + r0 + (tid & 15)) * 512, 0, 1);
    // 4) my wave's DMA done, then barrier so everyone's DMA+WMMA results are visible
    asm volatile("s_wait_asynccnt 0" ::: "memory");
    __syncthreads();

    // 5) gate nonlinearity + state update, emit hs
#pragma unroll
    for (int k = 0; k < 8; ++k) {
      int r = rbase + 2 * k;
      float gi = sGates[r][j]       + (float)sXg[r][j];
      float gf = sGates[r][128 + j] + (float)sXg[r][128 + j];
      float gg = sGates[r][256 + j] + (float)sXg[r][256 + j];
      float go = sGates[r][384 + j] + (float)sXg[r][384 + j];
      float iv = 1.f / (1.f + __expf(-gi));
      float fv = 1.f / (1.f + __expf(-gf));
      float gv = tanhf(gg);
      float ov = 1.f / (1.f + __expf(-go));
      float cv = fv * creg[k] + iv * gv;
      float hv = ov * tanhf(cv);
      creg[k] = cv;
      hreg[k] = hv;
      hs[((size_t)t * BATCH + r0 + r) * HIDD + j] = (_Float16)hv;
    }
    __syncthreads();
  }
}

// ---------- heads: out[TB,19] = hs @ [Wa;Wc]^T + [ba;bc] ----------
__global__ __launch_bounds__(256)
void heads_kernel(const _Float16* __restrict__ hs, const _Float16* __restrict__ Whead,
                  const float* __restrict__ bhead, float* __restrict__ out) {
  const int lane = threadIdx.x & 31;
  const int wv = threadIdx.x >> 5;
  const int row0 = blockIdx.x * 128 + wv * 16;
  const int col = lane & 15;
  const int rb = (lane & 16) ? 8 : 0;
  v16h a[4];
#pragma unroll
  for (int kt = 0; kt < 4; ++kt)
    a[kt] = load_A_f16(hs + (size_t)row0 * HIDD + kt * 32, HIDD, lane);
#pragma unroll
  for (int nt = 0; nt < 2; ++nt) {
    v8f acc = {};
#pragma unroll
    for (int kt = 0; kt < 4; ++kt)
      acc = wmma_f16(a[kt], load_B_W_f16(Whead, HIDD, nt * 16, kt * 32, lane), acc);
    int gcol = nt * 16 + col;
    float bv = bhead[gcol];
#pragma unroll
    for (int v = 0; v < 8; ++v) {
      if (gcol < 19)
        out[(size_t)(row0 + rb + v) * 19 + gcol] = acc[v] + bv;
    }
  }
}

extern "C" void kernel_launch(void* const* d_in, const int* in_sizes, int n_in,
                              void* d_out, int out_size, void* d_ws, size_t ws_size,
                              hipStream_t stream) {
  (void)in_sizes; (void)n_in; (void)out_size; (void)ws_size;
  const float* x    = (const float*)d_in[0];
  const float* done = (const float*)d_in[1];
  const float* h0   = (const float*)d_in[2];
  const float* c0   = (const float*)d_in[3];
  const float* W1   = (const float*)d_in[4];
  const float* b1   = (const float*)d_in[5];
  const float* W2   = (const float*)d_in[6];
  const float* b2   = (const float*)d_in[7];
  const float* W3   = (const float*)d_in[8];
  const float* b3   = (const float*)d_in[9];
  const float* Wih  = (const float*)d_in[10];
  const float* Whh  = (const float*)d_in[11];
  const float* bih  = (const float*)d_in[12];
  const float* bhh  = (const float*)d_in[13];
  const float* Wa   = (const float*)d_in[14];
  const float* ba   = (const float*)d_in[15];
  const float* Wc   = (const float*)d_in[16];
  const float* bc   = (const float*)d_in[17];
  float* out = (float*)d_out;

  char* ws = (char*)d_ws;
  size_t off = 0;
  _Float16* hid   = (_Float16*)(ws + off); off += (size_t)TB * LSD * 2;   // 64MB
  _Float16* xg    = (_Float16*)(ws + off); off += (size_t)TB * 512 * 2;   // 256MB
  _Float16* hs    = (_Float16*)(ws + off); off += (size_t)TB * HIDD * 2;  // 64MB
  _Float16* Whh16 = (_Float16*)(ws + off); off += (size_t)512 * 128 * 2;
  _Float16* Wih16 = (_Float16*)(ws + off); off += (size_t)512 * 128 * 2;
  _Float16* W1h   = (_Float16*)(ws + off); off += (size_t)128 * 128 * 2;
  _Float16* W2h   = (_Float16*)(ws + off); off += (size_t)32 * 128 * 2;
  _Float16* W3h   = (_Float16*)(ws + off); off += (size_t)128 * 32 * 2;
  _Float16* Whead = (_Float16*)(ws + off); off += (size_t)32 * 128 * 2;
  float*    bhead = (float*)(ws + off);    off += 32 * 4;

  prep_kernel<<<256, 256, 0, stream>>>(Whh, Whh16, W1, W1h, W2, W2h, W3, W3h,
                                       Wih, Wih16, Wa, ba, Wc, bc, Whead, bhead);
  encoder_kernel<<<TB / 128, 256, 0, stream>>>(x, W1h, b1, W2h, b2, W3h, b3, hid);
  xgates_kernel<<<dim3(TB / 128, 4), 256, 0, stream>>>(hid, Wih16, bih, bhh, xg);
  lstm_kernel<<<BATCH / 16, 256, 0, stream>>>(h0, c0, done, Whh16, xg, hs);
  heads_kernel<<<TB / 128, 256, 0, stream>>>(hs, Whead, bhead, out);
}
